// BinaryConv_11819749998891
// MI455X (gfx1250) — 8-bit matrix operand order repaired and checked by reading, compiled, not run on hardware
//
#include <hip/hip_runtime.h>


typedef int   v16i __attribute__((ext_vector_type(16)));
typedef float v8f  __attribute__((ext_vector_type(8)));

#define HH   224
#define WW   224
#define HWSZ 50176
#define NB   16
#define NCH  64
#define OCH  64
#define KTOT 576
#define KPAD 640
#define MTILE 128
#define A_STRIDE 672
#define OT_STRIDE 132

#if defined(__gfx1250__) && \
    __has_builtin(__builtin_amdgcn_global_load_async_to_lds_b32) && \
    __has_builtin(__builtin_amdgcn_s_wait_asynccnt)
#define HAVE_ASYNC_LDS 1
typedef __attribute__((address_space(1))) int* gbl_i32p;
typedef __attribute__((address_space(3))) int* lds_i32p;
#else
#define HAVE_ASYNC_LDS 0
#endif

__device__ __forceinline__ unsigned int sign_fp8(float v) {
    return v > 0.f ? 0x38u : (v < 0.f ? 0xB8u : 0x00u);
}

__global__ void wprep_kernel(const float* __restrict__ w,
                             float* __restrict__ scale,
                             unsigned int* __restrict__ bt  ) {
    __shared__ float red[64];
    const int o = blockIdx.x;
    const int t = threadIdx.x;
    const float* wo = w + o * KTOT;

    float s = 0.f;
    #pragma unroll
    for (int j = 0; j < 9; ++j) s += fabsf(wo[t * 9 + j]);
    red[t] = s;
    __syncthreads();
    if (t == 0) {
        float tot = 0.f;
        for (int i = 0; i < 64; ++i) tot += red[i];
        scale[o] = tot * (1.0f / 576.0f);
    }
    for (int j = 0; j < 3; ++j) {
        int d = j * 64 + t;
        if (d < 160) {
            unsigned int dw = 0;
            #pragma unroll
            for (int b = 0; b < 4; ++b) {
                int k = d * 4 + b;
                unsigned int byte = 0;
                if (k < KTOT) {
                    int i = k & 63, tap = k >> 6;
                    byte = sign_fp8(wo[i * 9 + tap]);
                }
                dw |= byte << (8 * b);
            }
            bt[o * 160 + d] = dw;
        }
    }
}

__global__ void __launch_bounds__(256)
spack_kernel(const float* __restrict__ x, unsigned int* __restrict__ xs32) {
    __shared__ unsigned char tile[64 * 68];
    const int tid = threadIdx.x;
    const unsigned int n     = blockIdx.x / (HWSZ / 64);
    const unsigned int hw0   = (blockIdx.x % (HWSZ / 64)) * 64;
    const float* xn = x + ((size_t)n * NCH * HWSZ + hw0);

    #pragma unroll
    for (int it = 0; it < 16; ++it) {
        int idx = it * 256 + tid;
        int c = idx >> 6, p = idx & 63;
        float v = __builtin_nontemporal_load(xn + (size_t)c * HWSZ + p);
        tile[p * 68 + c] = (unsigned char)sign_fp8(v);
    }
    __syncthreads();
    unsigned int* dst = xs32 + ((size_t)n * HWSZ + hw0) * 16;
    #pragma unroll
    for (int j = 0; j < 4; ++j) {
        int idx = j * 256 + tid;
        int p = idx >> 4, cd = idx & 15;
        dst[(size_t)p * 16 + cd] = *(const unsigned int*)(tile + p * 68 + cd * 4);
    }
}

template <bool PACKED>
__global__ void __launch_bounds__(256)
bconv_kernel(const float* __restrict__ x,
             const unsigned char* __restrict__ xs,
             const float* __restrict__ scale,
             const unsigned char* __restrict__ bt,
             float* __restrict__ out) {
    extern __shared__ unsigned char smem[];
    unsigned char* At = smem;

    const int tid = threadIdx.x;
    const unsigned int gp0 = blockIdx.x * (unsigned)MTILE;
    const unsigned int n   = gp0 / HWSZ;
    const unsigned int hw0 = gp0 % HWSZ;

    if (PACKED) {
        const unsigned char* xsn = xs + (size_t)n * HWSZ * 64;
        for (int it = 0; it < 72; ++it) {
            int idx = it * 256 + tid;
            int cd  = idx & 15;
            int t2  = idx >> 4;
            int tap = t2 % 9;
            int p   = t2 / 9;
            unsigned int hw = hw0 + p;
            int h = (int)(hw / WW), wc = (int)(hw % WW);
            int hh = h + tap / 3 - 1, ww2 = wc + tap % 3 - 1;
            unsigned char* dst = At + p * A_STRIDE + tap * 64 + cd * 4;
            bool inb = (hh >= 0 && hh < HH && ww2 >= 0 && ww2 < WW);
#if HAVE_ASYNC_LDS
            if (inb) {
                const unsigned char* src = xsn + ((size_t)hh * WW + ww2) * 64 + cd * 4;
                __builtin_amdgcn_global_load_async_to_lds_b32(
                    (gbl_i32p)src, (lds_i32p)dst, 0, 0);
            } else {
                *(unsigned int*)dst = 0u;
            }
#else
            unsigned int dw = 0;
            if (inb)
                dw = *(const unsigned int*)(xsn + ((size_t)hh * WW + ww2) * 64 + cd * 4);
            *(unsigned int*)dst = dw;
#endif
        }
    } else {
        const float* xn = x + (size_t)n * NCH * HWSZ;
        for (int it = 0; it < 72; ++it) {
            int idx = it * 256 + tid;
            int p   = idx & 127;
            int kg  = idx >> 7;
            int k0  = kg * 4;
            int tap = k0 >> 6;
            int i0  = k0 & 63;
            int kh  = tap / 3 - 1, kw = tap % 3 - 1;
            unsigned int hw = hw0 + p;
            int h = (int)(hw / WW), wc = (int)(hw % WW);
            int hh = h + kh, wcc = wc + kw;
            unsigned int dw = 0;
            if (hh >= 0 && hh < HH && wcc >= 0 && wcc < WW) {
                const float* px = xn + (size_t)i0 * HWSZ + (size_t)hh * WW + wcc;
                #pragma unroll
                for (int b = 0; b < 4; ++b)
                    dw |= sign_fp8(px[(size_t)b * HWSZ]) << (8 * b);
            }
            *(unsigned int*)(At + p * A_STRIDE + k0) = dw;
        }
    }
    for (int it = 0; it < 8; ++it) {
        int idx = it * 256 + tid;
        int p = idx >> 4, d = idx & 15;
        *(unsigned int*)(At + p * A_STRIDE + 576 + d * 4) = 0u;
    }
#if HAVE_ASYNC_LDS
    if (PACKED) __builtin_amdgcn_s_wait_asynccnt(0);
#endif
    __syncthreads();

    const int lane = tid & 31;
    const int wave = tid >> 5;
    const int row  = lane & 15;
    const int hi   = lane >> 4;
    const unsigned char* arow = At + (wave * 16 + row) * A_STRIDE;

    v8f acc[4] = {};
    #pragma unroll
    for (int c = 0; c < 5; ++c) {
        const int kbase = c * 128;
        v16i a;
        #pragma unroll
        for (int h2 = 0; h2 < 2; ++h2)
            #pragma unroll
            for (int q = 0; q < 4; ++q) {
                unsigned long long d = *(const unsigned long long*)
                    (arow + kbase + h2 * 64 + (q >> 1) * 32 + hi * 16 + (q & 1) * 8);
                a[h2 * 8 + q * 2]     = (int)(unsigned int)d;
                a[h2 * 8 + q * 2 + 1] = (int)(unsigned int)(d >> 32);
            }
        #pragma unroll
        for (int nt = 0; nt < 4; ++nt) {
            const unsigned char* brow = bt + (size_t)(nt * 16 + row) * KPAD + kbase + hi * 16;
            v16i b;
            #pragma unroll
            for (int cc = 0; cc < 4; ++cc) {
                const int4 q4 = *(const int4*)(brow + cc * 32);
                b[cc * 4 + 0] = q4.x; b[cc * 4 + 1] = q4.y;
                b[cc * 4 + 2] = q4.z; b[cc * 4 + 3] = q4.w;
            }
            acc[nt] = __builtin_amdgcn_wmma_f32_16x16x128_fp8_fp8(
                a, b, (short)0, acc[nt], false, false);
        }
    }

    __syncthreads();
    float* Ot = (float*)smem;
    #pragma unroll
    for (int nt = 0; nt < 4; ++nt) {
        const int o = nt * 16 + row;
        const float sc = scale[o];
        #pragma unroll
        for (int v = 0; v < 8; ++v) {
            int m = wave * 16 + v + hi * 8;
            Ot[o * OT_STRIDE + m] = acc[nt][v] * sc;
        }
    }
    __syncthreads();
    float* outn = out + (size_t)n * OCH * HWSZ + hw0;
    for (int j = 0; j < 32; ++j) {
        int idx = j * 256 + tid;
        int p = idx & 127, o = idx >> 7;
        __builtin_nontemporal_store(Ot[o * OT_STRIDE + p], outn + (size_t)o * HWSZ + p);
    }
}

extern "C" void kernel_launch(void* const* d_in, const int* in_sizes, int n_in,
                              void* d_out, int out_size, void* d_ws, size_t ws_size,
                              hipStream_t stream) {
    const float* x = (const float*)d_in[0];
    const float* w = (const float*)d_in[1];
    float* out = (float*)d_out;

    float*        scale = (float*)d_ws;
    unsigned int* bt    = (unsigned int*)((char*)d_ws + 256);
    const size_t  xs_off = 256 + 64 * KPAD;
    const size_t  xs_bytes = (size_t)NB * HWSZ * NCH;
    unsigned char* xs = (unsigned char*)d_ws + xs_off;

    wprep_kernel<<<64, 64, 0, stream>>>(w, scale, bt);

    const int nblocks = (NB * HWSZ) / MTILE;
    const size_t shmem = (size_t)MTILE * A_STRIDE;

    if (ws_size >= xs_off + xs_bytes) {
        spack_kernel<<<NB * (HWSZ / 64), 256, 0, stream>>>(x, (unsigned int*)xs);
        bconv_kernel<true><<<nblocks, 256, shmem, stream>>>(
            x, xs, scale, (const unsigned char*)bt, out);
    } else {
        bconv_kernel<false><<<nblocks, 256, shmem, stream>>>(
            x, xs, scale, (const unsigned char*)bt, out);
    }
}
